// NaiveDynMessage_54537494724735
// MI455X (gfx1250) — matrix operand order repaired and checked by reading, not compiled, not run on hardware
//
#include <hip/hip_runtime.h>
#include <hip/hip_bf16.h>

typedef __attribute__((ext_vector_type(16))) _Float16 v16h;
typedef __attribute__((ext_vector_type(8)))  float    v8f;

#define NV_ 100000
#define NE_ 300000
#define HV_ 128
#define HE_ 64
#define MV_ 128
#define ME_ 64

// K dims padded to multiples of 32 (WMMA K)
#define KN_RAW 198   // P+Q+HE+HV
#define KL_RAW 262   // HV+P+Q+HV
#define KN_ 224
#define KL_ 288
#define KN_CH 7      // 224/32
#define KL_CH 9      // 288/32
#define NT_N 8       // 128/16 output col tiles (neighbor)
#define NT_L 4       // 64/16  output col tiles (link)

// LDS row strides (halves), padded to break 64-bank aliasing across 16 rows
#define A1_STRIDE 228
#define A3_STRIDE 292

// d_ws layout in halves: swizzled B fragments
#define WN_SW_HALVES (KN_CH * NT_N * 32 * 16)   // 28672 halves = 57344 B
#define WL_SW_HALVES (KL_CH * NT_L * 32 * 16)   // 18432 halves = 36864 B

// ---------------------------------------------------------------------------
// Kernel 0: zero scatter-target + build f16 swizzled weight fragments in ws.
// B fragment layout used here, contiguous k, not ISA table order (32x16 B tile, wave32):
//   lane 0..15  -> K = 0..15   (half h -> K=h),        N = lane
//   lane 16..31 -> K = 16..31  (half h -> K=16+h),     N = lane-16
// stored so each lane reads one contiguous v16h (32B).
// ---------------------------------------------------------------------------
__global__ __launch_bounds__(256)
void prep_kernel(const float* __restrict__ Wn, const float* __restrict__ Wl,
                 float* __restrict__ mv,
                 _Float16* __restrict__ wn_sw, _Float16* __restrict__ wl_sw)
{
    const long i = (long)blockIdx.x * blockDim.x + threadIdx.x;

    if (i < (long)NV_ * MV_) mv[i] = 0.0f;

    if (i < WN_SW_HALVES) {
        const int h    = (int)(i & 15);
        const int lane = (int)((i >> 4) & 31);
        const long ct  = i >> 9;
        const int t    = (int)(ct & (NT_N - 1));
        const int c    = (int)(ct >> 3);
        const int K    = c * 32 + ((lane & 16) ? 16 : 0) + h;
        const int N    = t * 16 + (lane & 15);
        wn_sw[i] = (_Float16)((K < KN_RAW) ? Wn[(long)K * MV_ + N] : 0.0f);
    }
    if (i < WL_SW_HALVES) {
        const int h    = (int)(i & 15);
        const int lane = (int)((i >> 4) & 31);
        const long ct  = i >> 9;
        const int t    = (int)(ct & (NT_L - 1));
        const int c    = (int)(ct >> 2);
        const int K    = c * 32 + ((lane & 16) ? 16 : 0) + h;
        const int N    = t * 16 + (lane & 15);
        wl_sw[i] = (_Float16)((K < KL_RAW) ? Wl[(long)K * ME_ + N] : 0.0f);
    }
}

// A fragment (16x32 f16, wave32) in contiguous k order, matching B:
//   lane<16: pairs p=0..7 -> K = 2p;   lane>=16: +16.
__device__ __forceinline__ v16h load_a_frag(const _Float16* row, int c, int hiOff)
{
    union { v16h h; unsigned int u[8]; } r;
#pragma unroll
    for (int p = 0; p < 8; ++p) {
        const int k = c * 32 + 2 * p + 2 * hiOff;
        r.u[p] = *(const unsigned int*)(row + k);   // k even -> 4B aligned
    }
    return r.h;
}

__device__ __forceinline__ v8f wmma_f16(v16h a, v16h b, v8f c)
{
    return __builtin_amdgcn_wmma_f32_16x16x32_f16(false, a, false, b,
                                                  (short)0, c, false, false);
}

// ---------------------------------------------------------------------------
// Kernel 1: 16 edges per block, 4 waves. Stage concatenated A rows in LDS
// (f16), run 16x16x32 f16 WMMA against L2-resident swizzled weights, fused
// bias + LeakyReLU epilogue, scatter-add n1/n2 into mv, store me.
// ---------------------------------------------------------------------------
__global__ __launch_bounds__(128)
void edge_kernel(const float* __restrict__ hv, const float* __restrict__ he,
                 const float* __restrict__ pf, const float* __restrict__ qf,
                 const float* __restrict__ bn, const float* __restrict__ bl,
                 const int* __restrict__ src, const int* __restrict__ dst,
                 const _Float16* __restrict__ wn_sw,
                 const _Float16* __restrict__ wl_sw,
                 float* __restrict__ mv, float* __restrict__ me)
{
    __shared__ _Float16 A1[16 * A1_STRIDE];  // [-dp,-dq,he,rv2] padded to 224
    __shared__ _Float16 A2[16 * A1_STRIDE];  // [ dp, dq,he,rv1] padded to 224
    __shared__ _Float16 A3[16 * A3_STRIDE];  // [rv1,dp,dq,rv2]  padded to 288
    __shared__ int sSrc[16];
    __shared__ int sDst[16];

    const int  tid = threadIdx.x;
    const long e0  = (long)blockIdx.x * 16;

    // ---------------- stage features: 8 threads per edge ----------------
    {
        const int  le    = tid >> 3;
        const int  sub   = tid & 7;
        const long e     = e0 + le;
        const bool valid = (e < NE_);
        const int  s = valid ? src[e] : 0;
        const int  d = valid ? dst[e] : 0;
        _Float16* a1 = A1 + le * A1_STRIDE;
        _Float16* a2 = A2 + le * A1_STRIDE;
        _Float16* a3 = A3 + le * A3_STRIDE;

        if (sub == 0) {
            sSrc[le] = s; sDst[le] = d;
#pragma unroll
            for (int k = 0; k < 3; ++k) {
                const float dp = valid ? (pf[s * 3 + k] - pf[d * 3 + k]) : 0.0f;
                const float dq = valid ? (qf[s * 3 + k] - qf[d * 3 + k]) : 0.0f;
                a1[k]     = (_Float16)(-dp);  a1[3 + k]   = (_Float16)(-dq);
                a2[k]     = (_Float16)( dp);  a2[3 + k]   = (_Float16)( dq);
                a3[128+k] = (_Float16)( dp);  a3[131 + k] = (_Float16)( dq);
            }
        }
        for (int j = sub; j < HE_; j += 8) {
            const _Float16 h = (_Float16)(valid ? he[e * HE_ + j] : 0.0f);
            a1[6 + j] = h; a2[6 + j] = h;
        }
        for (int j = sub; j < HV_; j += 8) {
            const _Float16 hs = (_Float16)(valid ? hv[(long)s * HV_ + j] : 0.0f);
            const _Float16 hd = (_Float16)(valid ? hv[(long)d * HV_ + j] : 0.0f);
            a1[70 + j]  = hd;   // rv2
            a2[70 + j]  = hs;   // rv1
            a3[j]       = hs;   // rv1
            a3[134 + j] = hd;   // rv2
        }
        for (int j = KN_RAW + sub; j < KN_; j += 8) {
            a1[j] = (_Float16)0.0f; a2[j] = (_Float16)0.0f;
        }
        for (int j = KL_RAW + sub; j < KL_; j += 8) a3[j] = (_Float16)0.0f;
    }
    __syncthreads();

    // ---------------- matrix phase (uniform control flow, EXEC all-1) ----
    const int wave  = tid >> 5;
    const int lane  = tid & 31;
    const int mlane = lane & 15;
    const int hiOff = (lane & 16) ? 8 : 0;

    const _Float16* r1 = A1 + mlane * A1_STRIDE;
    const _Float16* r2 = A2 + mlane * A1_STRIDE;
    const _Float16* r3 = A3 + mlane * A3_STRIDE;

    v8f zero = {};
    v8f accn1[2] = { zero, zero };
    v8f accn2[2] = { zero, zero };
    v8f accme    = zero;

    // neighbor messages: two N-tiles per wave, shared B between n1/n2
#pragma unroll
    for (int c = 0; c < KN_CH; ++c) {
        const v16h a1f = load_a_frag(r1, c, hiOff);
        const v16h a2f = load_a_frag(r2, c, hiOff);
#pragma unroll
        for (int tt = 0; tt < 2; ++tt) {
            const int t = wave * 2 + tt;
            const v16h b = *(const v16h*)(wn_sw + (((long)(c * NT_N + t)) * 32 + lane) * 16);
            accn1[tt] = wmma_f16(a1f, b, accn1[tt]);
            accn2[tt] = wmma_f16(a2f, b, accn2[tt]);
        }
    }
    // link messages: one N-tile per wave
#pragma unroll
    for (int c = 0; c < KL_CH; ++c) {
        const v16h a3f = load_a_frag(r3, c, hiOff);
        const v16h b = *(const v16h*)(wl_sw + (((long)(c * NT_L + wave)) * 32 + lane) * 16);
        accme = wmma_f16(a3f, b, accme);
    }

    // ---------------- epilogue: C/D layout -> bias + LeakyReLU ----------
    const int mbase = (lane & 16) ? 8 : 0;
#pragma unroll
    for (int tt = 0; tt < 2; ++tt) {
        const int col   = (wave * 2 + tt) * 16 + mlane;
        const float bias = bn[col];
#pragma unroll
        for (int r = 0; r < 8; ++r) {
            const int  m = mbase + r;
            const long e = e0 + m;
            if (e < NE_) {
                float x = accn1[tt][r] + bias;
                x = x > 0.0f ? x : 0.01f * x;
                atomicAdd(mv + (long)sSrc[m] * MV_ + col, x);
                float y = accn2[tt][r] + bias;
                y = y > 0.0f ? y : 0.01f * y;
                atomicAdd(mv + (long)sDst[m] * MV_ + col, y);
            }
        }
    }
    {
        const int colL   = wave * 16 + mlane;
        const float bias = bl[colL];
#pragma unroll
        for (int r = 0; r < 8; ++r) {
            const int  m = mbase + r;
            const long e = e0 + m;
            if (e < NE_) {
                float x = accme[r] + bias;
                x = x > 0.0f ? x : 0.01f * x;
                me[e * ME_ + colL] = x;
            }
        }
    }
}

// ---------------------------------------------------------------------------
// Kernel 2: in-place ELU over aggregated vertex features.
// ---------------------------------------------------------------------------
__global__ __launch_bounds__(256)
void elu_kernel(float* __restrict__ mv)
{
    const long i = (long)blockIdx.x * blockDim.x + threadIdx.x;
    if (i < (long)NV_ * MV_) {
        const float x = mv[i];
        mv[i] = x > 0.0f ? x : (__expf(x) - 1.0f);
    }
}

extern "C" void kernel_launch(void* const* d_in, const int* in_sizes, int n_in,
                              void* d_out, int out_size, void* d_ws, size_t ws_size,
                              hipStream_t stream)
{
    const float* hv = (const float*)d_in[0];
    const float* he = (const float*)d_in[1];
    const float* pf = (const float*)d_in[2];
    const float* qf = (const float*)d_in[3];
    const float* Wn = (const float*)d_in[4];
    const float* bn = (const float*)d_in[5];
    const float* Wl = (const float*)d_in[6];
    const float* bl = (const float*)d_in[7];
    const int*  src = (const int*)d_in[8];
    const int*  dst = (const int*)d_in[9];

    float* mv = (float*)d_out;                       // [NV, 128]
    float* me = mv + (long)NV_ * MV_;                // [NE, 64]

    _Float16* wn_sw = (_Float16*)d_ws;               // 57344 B
    _Float16* wl_sw = wn_sw + WN_SW_HALVES;          // 36864 B  (total ~94 KB)

    const long mvElems = (long)NV_ * MV_;
    const int  prepGrid = (int)((mvElems + 255) / 256);
    prep_kernel<<<prepGrid, 256, 0, stream>>>(Wn, Wl, mv, wn_sw, wl_sw);

    const int edgeGrid = (NE_ + 15) / 16;            // 18750 blocks, 16 edges each
    edge_kernel<<<edgeGrid, 128, 0, stream>>>(hv, he, pf, qf, bn, bl, src, dst,
                                              wn_sw, wl_sw, mv, me);

    elu_kernel<<<prepGrid, 256, 0, stream>>>(mv);
}
